// ChaosSSMCore_2774548873892
// MI455X (gfx1250) — compile-verified, matrix operand order checked by reading, not run on hardware
//
#include <hip/hip_runtime.h>
#include <hip/hip_bf16.h>
#include <stdint.h>

// ---------------------------------------------------------------------------
// Types for CDNA5 WMMA (wave32): A/B = 16 bf16 (8 VGPRs), C/D = 8 f32.
// ---------------------------------------------------------------------------
typedef __bf16 bf16x16 __attribute__((ext_vector_type(16)));
typedef float  f32x8   __attribute__((ext_vector_type(8)));

union Frag {
  bf16x16 v;
  uint4   u[2];   // two 16-byte halves -> ds_load_b128 / global_load_b128
};

__device__ __forceinline__ uint32_t f2bf(float f) {
  uint32_t u = __float_as_uint(f);
  return ((u + 0x7FFFu + ((u >> 16) & 1u)) >> 16) & 0xFFFFu;  // RNE
}
__device__ __forceinline__ float bf2f(uint32_t b) {
  return __uint_as_float((b & 0xFFFFu) << 16);
}

#define BATCH 128
#define SEQ   2048
#define DIM   256
#define NTOK  (BATCH * SEQ)     // 262144 tokens
#define ROWS_PER_WG 128
#define LDS_STRIDE 264          // 256 + 8 bf16 pad -> conflict-free b128 frag reads
#define NCHUNK 32
#define CHUNK  64               // SEQ / NCHUNK

// ---------------------------------------------------------------------------
// Kernel 0: weights fp32 -> bf16 (L2-resident thereafter), ea = exp(log_a).
// mat order: 0=W_delta 1=W_in 2=W_select 3=W_gate 4=W_out
// ---------------------------------------------------------------------------
__global__ __launch_bounds__(256)
void wcvt_kernel(const float* __restrict__ Wd, const float* __restrict__ Wi,
                 const float* __restrict__ Wsel, const float* __restrict__ Wg,
                 const float* __restrict__ Wo, const float* __restrict__ log_a,
                 unsigned short* __restrict__ Wb, float* __restrict__ ea) {
  const int mat = blockIdx.y;
  const int i = blockIdx.x * 256 + threadIdx.x;   // 65536 per matrix
  const float* src = (mat == 0) ? Wd : (mat == 1) ? Wi : (mat == 2) ? Wsel
                   : (mat == 3) ? Wg : Wo;
  Wb[mat * 65536 + i] = (unsigned short)f2bf(src[i]);
  if (mat == 4 && blockIdx.x == 0) ea[threadIdx.x] = __expf(log_a[threadIdx.x]);
}

// ---------------------------------------------------------------------------
// Kernel 1: fused 4-way projection GEMM + activations.
//   delta = softplus(x@Wd^T); decay = exp(-delta*ea); sel = sigmoid(x@Ws^T)
//   update = delta*sel*(x@Wi^T); gate = silu(x@Wg^T)
// Writes du = pack(bf16(decay), bf16(update)) and gate bf16.
// One 128-row tile per WG; 8 waves; 16 (16x16) output tiles per wave;
// each tile: 8 K-steps x 4 WMMAs sharing one A fragment.
// ---------------------------------------------------------------------------
__global__ __launch_bounds__(256)
void proj_kernel(const float* __restrict__ x,
                 const unsigned short* __restrict__ Wb,
                 const float* __restrict__ ea,
                 uint32_t* __restrict__ du,
                 unsigned short* __restrict__ gate) {
  __shared__ unsigned short xs[ROWS_PER_WG * LDS_STRIDE];
  const int tid = threadIdx.x;
  const int rowBase = blockIdx.x * ROWS_PER_WG;

  // Stage x tile fp32 -> bf16 into LDS (coalesced float4 reads, b64 LDS stores)
  for (int it = 0; it < 32; ++it) {
    const int i = it * 256 + tid;          // 8192 groups of 4 floats
    const int row = i >> 6;
    const int cg = i & 63;
    const float4 xv = *(const float4*)(x + (size_t)(rowBase + row) * DIM + cg * 4);
    uint2 p;
    p.x = f2bf(xv.x) | (f2bf(xv.y) << 16);
    p.y = f2bf(xv.z) | (f2bf(xv.w) << 16);
    *(uint2*)(&xs[row * LDS_STRIDE + cg * 4]) = p;
  }
  __syncthreads();

  const int wid = tid >> 5;
  const int lane = tid & 31;
  const int lrow = lane & 15;
  const int khi = lane >> 4;       // lanes 16..31 take the K+8 / K+24 halves
  const int koff = khi * 8;

  for (int tile = wid; tile < 128; tile += 8) {
    const int rowT = tile >> 4;    // 0..7
    const int colT = tile & 15;    // 0..15
    f32x8 c0 = {}, c1 = {}, c2 = {}, c3 = {};
    const int abase = (rowT * 16 + lrow) * LDS_STRIDE + koff;
    const int bbase = (colT * 16 + lrow) * DIM + koff;
#pragma unroll
    for (int k0 = 0; k0 < DIM; k0 += 32) {
      Frag a, b0, b1, b2, b3;
      const unsigned short* pa = &xs[abase + k0];
      a.u[0] = *(const uint4*)(pa);
      a.u[1] = *(const uint4*)(pa + 16);
      const unsigned short* pw = Wb + bbase + k0;
      b0.u[0] = *(const uint4*)(pw);
      b0.u[1] = *(const uint4*)(pw + 16);
      b1.u[0] = *(const uint4*)(pw + 65536);
      b1.u[1] = *(const uint4*)(pw + 65536 + 16);
      b2.u[0] = *(const uint4*)(pw + 131072);
      b2.u[1] = *(const uint4*)(pw + 131072 + 16);
      b3.u[0] = *(const uint4*)(pw + 196608);
      b3.u[1] = *(const uint4*)(pw + 196608 + 16);
      c0 = __builtin_amdgcn_wmma_f32_16x16x32_bf16(false, a.v, false, b0.v, (short)0, c0, false, false);
      c1 = __builtin_amdgcn_wmma_f32_16x16x32_bf16(false, a.v, false, b1.v, (short)0, c1, false, false);
      c2 = __builtin_amdgcn_wmma_f32_16x16x32_bf16(false, a.v, false, b2.v, (short)0, c2, false, false);
      c3 = __builtin_amdgcn_wmma_f32_16x16x32_bf16(false, a.v, false, b3.v, (short)0, c3, false, false);
    }
    // Epilogue in C-layout: c[j] @ (M = rowT*16 + j + 8*khi, N = colT*16 + lrow)
    const int col = colT * 16 + lrow;
    const float eav = ea[col];
    const size_t rb = (size_t)(rowBase + rowT * 16 + khi * 8);
#pragma unroll
    for (int j = 0; j < 8; ++j) {
      const float dv = c0[j], uv = c1[j], sv = c2[j], gv = c3[j];
      const float dlt = fmaxf(dv, 0.f) + __logf(1.f + __expf(-fabsf(dv)));  // softplus
      const float dec = __expf(-dlt * eav);
      const float sig = 1.f / (1.f + __expf(-sv));
      const float upd = dlt * sig * uv;
      const float gtv = gv / (1.f + __expf(-gv));                           // silu
      const size_t idx = (rb + j) * DIM + col;
      du[idx] = f2bf(dec) | (f2bf(upd) << 16);
      gate[idx] = (unsigned short)f2bf(gtv);
    }
  }
}

// ---------------------------------------------------------------------------
// Kernel 2: chunked scan pass 1 — per (b, chunk, d): P = prod decay, S = local scan end.
// ---------------------------------------------------------------------------
__global__ __launch_bounds__(256)
void scan1_kernel(const uint32_t* __restrict__ du,
                  float* __restrict__ Pc, float* __restrict__ Sc) {
  const int b = blockIdx.x, c = blockIdx.y, d = threadIdx.x;
  size_t base = ((size_t)b * SEQ + (size_t)c * CHUNK) * DIM + d;
  float P = 1.f, S = 0.f;
#pragma unroll 4
  for (int t = 0; t < CHUNK; ++t) {
    const uint32_t w = du[base + (size_t)t * DIM];
    const float dec = bf2f(w), upd = bf2f(w >> 16);
    S = fmaf(dec, S, upd);
    P *= dec;
  }
  const int s = (b * NCHUNK + c) * DIM + d;
  Pc[s] = P;
  Sc[s] = S;
}

// ---------------------------------------------------------------------------
// Kernel 3: scan pass 2 — rebuild carry from chunk summaries, replay local
// scan seeded with carry, emit y = state * gate (bf16).
// ---------------------------------------------------------------------------
__global__ __launch_bounds__(256)
void scan2_kernel(const uint32_t* __restrict__ du,
                  const unsigned short* __restrict__ gate,
                  const float* __restrict__ Pc, const float* __restrict__ Sc,
                  unsigned short* __restrict__ y) {
  const int b = blockIdx.x, c = blockIdx.y, d = threadIdx.x;
  float carry = 0.f;
  for (int cc = 0; cc < c; ++cc) {
    const int s = (b * NCHUNK + cc) * DIM + d;
    carry = fmaf(Pc[s], carry, Sc[s]);
  }
  float state = carry;
  size_t base = ((size_t)b * SEQ + (size_t)c * CHUNK) * DIM + d;
#pragma unroll 4
  for (int t = 0; t < CHUNK; ++t) {
    const size_t idx = base + (size_t)t * DIM;
    const uint32_t w = du[idx];
    state = fmaf(bf2f(w), state, bf2f(w >> 16));
    const float g = bf2f(gate[idx]);
    y[idx] = (unsigned short)f2bf(state * g);
  }
}

// ---------------------------------------------------------------------------
// Kernel 4: out = y @ W_out^T  (y bf16 from ws, fp32 accumulate/output).
// ---------------------------------------------------------------------------
__global__ __launch_bounds__(256)
void outgemm_kernel(const unsigned short* __restrict__ y,
                    const unsigned short* __restrict__ Wb,
                    float* __restrict__ out) {
  __shared__ unsigned short ys[ROWS_PER_WG * LDS_STRIDE];
  const int tid = threadIdx.x;
  const int rowBase = blockIdx.x * ROWS_PER_WG;
  for (int it = 0; it < 16; ++it) {
    const int i = it * 256 + tid;       // 4096 groups of 8 bf16
    const int row = i >> 5;
    const int cg = i & 31;
    *(uint4*)(&ys[row * LDS_STRIDE + cg * 8]) =
        *(const uint4*)(y + (size_t)(rowBase + row) * DIM + cg * 8);
  }
  __syncthreads();

  const int wid = tid >> 5, lane = tid & 31;
  const int lrow = lane & 15, khi = lane >> 4, koff = khi * 8;
  const unsigned short* Wo = Wb + 4 * 65536;

  for (int tile = wid; tile < 128; tile += 8) {
    const int rowT = tile >> 4;
    const int colT = tile & 15;
    f32x8 c = {};
    const int abase = (rowT * 16 + lrow) * LDS_STRIDE + koff;
    const int bbase = (colT * 16 + lrow) * DIM + koff;
#pragma unroll
    for (int k0 = 0; k0 < DIM; k0 += 32) {
      Frag a, b;
      const unsigned short* pa = &ys[abase + k0];
      a.u[0] = *(const uint4*)(pa);
      a.u[1] = *(const uint4*)(pa + 16);
      const unsigned short* pw = Wo + bbase + k0;
      b.u[0] = *(const uint4*)(pw);
      b.u[1] = *(const uint4*)(pw + 16);
      c = __builtin_amdgcn_wmma_f32_16x16x32_bf16(false, a.v, false, b.v, (short)0, c, false, false);
    }
    const int col = colT * 16 + lrow;
    const size_t rb = (size_t)(rowBase + rowT * 16 + khi * 8);
#pragma unroll
    for (int j = 0; j < 8; ++j) out[(rb + j) * DIM + col] = c[j];
  }
}

// ---------------------------------------------------------------------------
// Host launch. Workspace layout (bytes):
//   [0)                 Wb   : 5*65536 bf16            (640 KB)
//   [+1KB)              ea   : 256 f32
//   [..)                Pc,Sc: 128*32*256 f32 each     (4 MB each)
//   [..)                du   : N*256 u32 {decay,update}(256 MB)
//   [..)                gate : N*256 bf16              (128 MB)
//   [..)                y    : N*256 bf16              (128 MB)
// ---------------------------------------------------------------------------
extern "C" void kernel_launch(void* const* d_in, const int* in_sizes, int n_in,
                              void* d_out, int out_size, void* d_ws, size_t ws_size,
                              hipStream_t stream) {
  (void)in_sizes; (void)n_in; (void)out_size; (void)ws_size;
  const float* x    = (const float*)d_in[0];
  const float* Wi   = (const float*)d_in[1];
  const float* Wsel = (const float*)d_in[2];
  const float* Wg   = (const float*)d_in[3];
  const float* Wo   = (const float*)d_in[4];
  const float* Wd   = (const float*)d_in[5];
  const float* loga = (const float*)d_in[6];
  float* out = (float*)d_out;

  char* ws = (char*)d_ws;
  unsigned short* Wb = (unsigned short*)ws;
  float* ea = (float*)(ws + (size_t)5 * 65536 * 2);
  float* Pc = (float*)(ws + (size_t)5 * 65536 * 2 + 1024);
  float* Sc = Pc + (size_t)BATCH * NCHUNK * DIM;
  uint32_t* du = (uint32_t*)(Sc + (size_t)BATCH * NCHUNK * DIM);
  unsigned short* gate = (unsigned short*)(du + (size_t)NTOK * DIM);
  unsigned short* y = gate + (size_t)NTOK * DIM;

  wcvt_kernel<<<dim3(256, 5), 256, 0, stream>>>(Wd, Wi, Wsel, Wg, Wo, loga, Wb, ea);
  proj_kernel<<<NTOK / ROWS_PER_WG, 256, 0, stream>>>(x, Wb, ea, du, gate);
  scan1_kernel<<<dim3(BATCH, NCHUNK), 256, 0, stream>>>(du, Pc, Sc);
  scan2_kernel<<<dim3(BATCH, NCHUNK), 256, 0, stream>>>(du, gate, Pc, Sc, y);
  outgemm_kernel<<<NTOK / ROWS_PER_WG, 256, 0, stream>>>(y, Wb, out);
}
